// HEGCRCNN_1391569404152
// MI455X (gfx1250) — compile-verified, matrix operand order checked by reading, not run on hardware
//
#include <hip/hip_runtime.h>

#ifndef __has_builtin
#define __has_builtin(x) 0
#endif

#if defined(__HIP_DEVICE_COMPILE__)
#if __has_builtin(__builtin_amdgcn_global_load_async_to_lds_b128) && \
    __has_builtin(__builtin_amdgcn_s_wait_asynccnt)
#define ASYNC_LDS 1
#endif
#endif

typedef __attribute__((ext_vector_type(16))) _Float16 v16h;
typedef __attribute__((ext_vector_type(8)))  _Float16 v8h;
typedef __attribute__((ext_vector_type(8)))  float    v8f;

#define B_    64
#define HIMG  34
#define WIMG  51
#define CIN   50
#define C1    64
#define HP    17
#define WW    49
#define LH    128
#define NSEQ  (B_*HP)        // 1088
#define KCAT  192
#define NGATE 512
#define CAPS  256
#define C_IH  HP
#define C_IW  WW
#define C_IC  LH
#define C_OH  9
#define C_OW  41
#define C_K   (9*9*C_IC)     // 10368
#define C_M   (B_*C_OH*C_OW) // 23616
#define P_IC  256
#define P_K   (9*9*P_IC)     // 20736
#define P_OW  17
#define P_M   (B_*P_OW)      // 1088
#define NPRIM 8
#define PDIM  32
#define PUNIT 544
#define NOUT  103
#define ODIM  16

__device__ __forceinline__ float sigm(float v) { return 1.f / (1.f + __expf(-v)); }

#ifdef ASYNC_LDS
typedef int v4i_gcc __attribute__((vector_size(16)));
typedef __attribute__((address_space(1))) v4i_gcc* gv4i_p;
typedef __attribute__((address_space(3))) v4i_gcc* lv4i_p;
__device__ __forceinline__ void async_cp_b128(const _Float16* g, _Float16* l)
{
    __builtin_amdgcn_global_load_async_to_lds_b128(
        (gv4i_p)(void*)(const_cast<_Float16*>(g)),
        (lv4i_p)(void*)l, 0, 0);
}
#endif

// ---------------- conv1 + relu + H-maxpool, write seq fp16 [s][w][c] ----------------
__global__ __launch_bounds__(256) void conv1_pool(const float* __restrict__ x,
                                                  const float* __restrict__ cw,
                                                  const float* __restrict__ cb,
                                                  _Float16* __restrict__ seq)
{
    __shared__ float xt[2 * CIN * WIMG];   // [hr][ci][w]
    __shared__ float wt[C1 * CIN * 3];     // [c][ci][kw]
    int s = blockIdx.x;                    // b*17+hp
    int b = s / HP, hp = s - b * HP;
    int tid = threadIdx.x;
    for (int f = tid; f < 2 * CIN * WIMG; f += 256) {
        int hr = f / (CIN * WIMG);
        int r  = f - hr * (CIN * WIMG);
        int ci = r / WIMG;
        int w  = r - ci * WIMG;
        xt[f] = x[((size_t)(b * CIN + ci) * HIMG + (2 * hp + hr)) * WIMG + w];
    }
    for (int f = tid; f < C1 * CIN * 3; f += 256) wt[f] = cw[f];
    __syncthreads();
    for (int idx = tid; idx < WW * C1; idx += 256) {
        int c = idx / WW, w = idx - (idx / WW) * WW;
        float a0 = 0.f, a1 = 0.f;
        for (int ci = 0; ci < CIN; ci++) {
            const float* wp = &wt[(c * CIN + ci) * 3];
            const float* x0 = &xt[ci * WIMG + w];
            const float* x1 = &xt[CIN * WIMG + ci * WIMG + w];
#pragma unroll
            for (int kw = 0; kw < 3; kw++) { a0 += wp[kw] * x0[kw]; a1 += wp[kw] * x1[kw]; }
        }
        float v = fmaxf(fmaxf(a0, a1) + cb[c], 0.f);
        seq[((size_t)s * WW + w) * C1 + c] = (_Float16)v;
    }
}

// ---------------- weight packing ----------------
__global__ void pack_lstm(const float* __restrict__ wih, const float* __restrict__ whh,
                          const float* __restrict__ bih, const float* __restrict__ bhh,
                          _Float16* __restrict__ Wcat, float* __restrict__ bsum)
{
    int idx = blockIdx.x * 256 + threadIdx.x;
    if (idx >= NGATE * KCAT) return;
    int n = idx / KCAT, k = idx - n * KCAT;
    float v = (k < C1) ? wih[n * C1 + k] : whh[n * LH + (k - C1)];
    Wcat[idx] = (_Float16)v;
    if (k == 0) bsum[n] = bih[n] + bhh[n];
}

__global__ void pack_conv(const float* __restrict__ src, _Float16* __restrict__ dst,
                          int OC, int IC, int KH, int KW)
{
    int total = OC * IC * KH * KW;
    int idx = blockIdx.x * 256 + threadIdx.x;
    if (idx >= total) return;
    int o  = idx / (IC * KH * KW);
    int r  = idx - o * (IC * KH * KW);
    int c  = r / (KH * KW);
    int r2 = r - c * (KH * KW);
    int kh = r2 / KW, kw = r2 - kh * KW;
    dst[(size_t)o * (KH * KW * IC) + (kh * KW + kw) * IC + c] = (_Float16)src[idx];
}

// ---------------- LSTM helpers ----------------
__global__ void lstm_init(float* __restrict__ cst, _Float16* __restrict__ xcat)
{
    int idx = blockIdx.x * 256 + threadIdx.x;
    if (idx >= NSEQ * LH) return;
    int s = idx >> 7, h = idx & 127;
    cst[idx] = 0.f;
    xcat[s * KCAT + C1 + h] = (_Float16)0.f;
}

__global__ void lstm_xcopy(const _Float16* __restrict__ seq, _Float16* __restrict__ xcat, int t)
{
    int idx = blockIdx.x * 256 + threadIdx.x;
    if (idx >= NSEQ * C1) return;
    int s = idx >> 6, c = idx & 63;
    xcat[s * KCAT + c] = seq[((size_t)s * WW + t) * C1 + c];
}

__global__ void lstm_gate(const float* __restrict__ gates, const float* __restrict__ bsum,
                          float* __restrict__ cst, _Float16* __restrict__ xcat,
                          _Float16* __restrict__ hs, int t)
{
    int idx = blockIdx.x * 256 + threadIdx.x;
    if (idx >= NSEQ * LH) return;
    int s = idx >> 7, h = idx & 127;
    const float* g = gates + (size_t)s * NGATE;
    float gi = g[h]          + bsum[h];
    float gf = g[LH + h]     + bsum[LH + h];
    float gg = g[2 * LH + h] + bsum[2 * LH + h];
    float go = g[3 * LH + h] + bsum[3 * LH + h];
    float c  = cst[idx];
    float cn = sigm(gf) * c + sigm(gi) * tanhf(gg);
    float hn = sigm(go) * tanhf(cn);
    cst[idx] = cn;
    xcat[s * KCAT + C1 + h] = (_Float16)hn;
    hs[((size_t)s * WW + t) * LH + h] = (_Float16)hn;
}

// ---------------- WMMA GEMM (plain or implicit-im2col) ----------------
// C[M,N] = A[M,K] * Bt[N,K]^T ; fp16 in, fp32 accum; async-LDS double buffered
template<bool CONV>
__global__ __launch_bounds__(256) void gemm_wmma(
    const _Float16* __restrict__ A, const _Float16* __restrict__ Bt,
    float* __restrict__ C32, _Float16* __restrict__ C16,
    const float* __restrict__ bias, int M, int N, int K, int relu,
    int IH, int IW, int IC, int KH, int KW, int SH, int SW, int OH, int OW)
{
    __shared__ _Float16 As[2][128][40];   // 32 + 8 pad
    __shared__ _Float16 Bs[2][128][40];
    const int tid  = threadIdx.x;
    const int m0   = blockIdx.x * 128;
    const int n0   = blockIdx.y * 128;
    const int lane = tid & 31;
    const int wid  = tid >> 5;
    const int wm   = (wid & 1) * 64;   // wave M offset
    const int wn   = (wid >> 1) * 32;  // wave N offset

    v8f acc[4][2];
#pragma unroll
    for (int i = 0; i < 4; i++)
#pragma unroll
        for (int jn = 0; jn < 2; jn++)
#pragma unroll
            for (int r = 0; r < 8; r++) acc[i][jn][r] = 0.f;

    const int lrow = tid >> 1;          // 0..127
    const int lks  = (tid & 1) << 4;    // 0 or 16

    // per-thread global source for the A tile at K offset k0
    auto a_src = [&](int k0) -> const _Float16* {
        int mg = m0 + lrow;
        if (CONV) {
            int ohw = OH * OW;
            int bb = mg / ohw;
            int r  = mg - bb * ohw;
            int oh = r / OW;
            int ow = r - oh * OW;
            int k  = k0 + lks;
            int tt = k / IC;
            int ic = k - tt * IC;
            int kw = tt % KW;
            int kh = tt / KW;
            return A + ((size_t)((bb * IH + oh * SH + kh) * IW + (ow * SW + kw))) * IC + ic;
        } else {
            return A + (size_t)mg * K + (k0 + lks);
        }
    };

    auto compute = [&](int buf) {
        const int lr   = lane & 15;
        const int kh16 = (lane >> 4) << 4;
        v16h af[4], bf[2];
#pragma unroll
        for (int mf = 0; mf < 4; mf++) {
            const _Float16* p = &As[buf][wm + mf * 16 + lr][kh16];
            v8h lo = *(const v8h*)p;
            v8h hi = *(const v8h*)(p + 8);
#pragma unroll
            for (int q = 0; q < 8; q++) { af[mf][q] = lo[q]; af[mf][q + 8] = hi[q]; }
        }
#pragma unroll
        for (int nf = 0; nf < 2; nf++) {
            const _Float16* p = &Bs[buf][wn + nf * 16 + lr][kh16];
            v8h lo = *(const v8h*)p;
            v8h hi = *(const v8h*)(p + 8);
#pragma unroll
            for (int q = 0; q < 8; q++) { bf[nf][q] = lo[q]; bf[nf][q + 8] = hi[q]; }
        }
#pragma unroll
        for (int mf = 0; mf < 4; mf++)
#pragma unroll
            for (int nf = 0; nf < 2; nf++)
                acc[mf][nf] = __builtin_amdgcn_wmma_f32_16x16x32_f16(
                    false, af[mf], false, bf[nf], (short)0, acc[mf][nf], false, false);
    };

#ifdef ASYNC_LDS
    const bool fullM = (m0 + 128 <= M);   // block-uniform
    if (fullM) {
        // async memory->LDS, double buffered: overlap next tile with WMMAs
        auto issue = [&](int k0, int buf) {
            const _Float16* sa = a_src(k0);
            _Float16* da = &As[buf][lrow][lks];
            async_cp_b128(sa, da);
            async_cp_b128(sa + 8, da + 8);
            const _Float16* sb = Bt + (size_t)(n0 + lrow) * K + (k0 + lks);
            _Float16* db = &Bs[buf][lrow][lks];
            async_cp_b128(sb, db);
            async_cp_b128(sb + 8, db + 8);
        };
        const int nk = K >> 5;
        issue(0, 0);
        for (int kk = 0; kk < nk; kk++) {
            const int buf = kk & 1;
            __builtin_amdgcn_s_wait_asynccnt(0);
            __syncthreads();                        // tile[buf] resident for all waves
            if (kk + 1 < nk) issue((kk + 1) << 5, buf ^ 1);
            compute(buf);
            __syncthreads();                        // done reading tile[buf]
        }
    } else
#endif
    {
        // synchronous fallback (partial-M edge tiles / no async builtin)
        for (int k0 = 0; k0 < K; k0 += 32) {
            __syncthreads();
            int mg = m0 + lrow;
            v8h a0, a1;
            if (mg < M) {
                const _Float16* src = a_src(k0);
                a0 = *(const v8h*)src;
                a1 = *(const v8h*)(src + 8);
            } else {
#pragma unroll
                for (int q = 0; q < 8; q++) { a0[q] = (_Float16)0.f; a1[q] = (_Float16)0.f; }
            }
            *(v8h*)&As[0][lrow][lks]     = a0;
            *(v8h*)&As[0][lrow][lks + 8] = a1;

            int ng = n0 + lrow;
            v8h b0, b1;
            if (ng < N) {
                const _Float16* srcb = Bt + (size_t)ng * K + (k0 + lks);
                b0 = *(const v8h*)srcb;
                b1 = *(const v8h*)(srcb + 8);
                __builtin_prefetch((const void*)(srcb + 32), 0, 0);
            } else {
#pragma unroll
                for (int q = 0; q < 8; q++) { b0[q] = (_Float16)0.f; b1[q] = (_Float16)0.f; }
            }
            *(v8h*)&Bs[0][lrow][lks]     = b0;
            *(v8h*)&Bs[0][lrow][lks + 8] = b1;
            __syncthreads();
            compute(0);
        }
    }

    const int lr   = lane & 15;
    const int hsel = lane >> 4;
#pragma unroll
    for (int mf = 0; mf < 4; mf++) {
#pragma unroll
        for (int nf = 0; nf < 2; nf++) {
            int col = n0 + wn + nf * 16 + lr;
            float bv = (bias && col < N) ? bias[col] : 0.f;
#pragma unroll
            for (int r = 0; r < 8; r++) {
                int row = m0 + wm + mf * 16 + hsel * 8 + r;
                if (row < M && col < N) {
                    float v = acc[mf][nf][r] + bv;
                    if (relu) v = fmaxf(v, 0.f);
                    if (C32) C32[(size_t)row * N + col] = v;
                    if (C16) C16[(size_t)row * N + col] = (_Float16)v;
                }
            }
        }
    }
}

// ---------------- squash primary capsules; write ut[i][b*8+p] ----------------
__global__ __launch_bounds__(256) void squash_u(const float* __restrict__ praw,
                                                float* __restrict__ ut)
{
    __shared__ float red[256];
    int bp = blockIdx.x;          // b*8+p
    int b = bp >> 3, p = bp & 7;
    int tid = threadIdx.x;
    float vals[3]; float ss = 0.f;
#pragma unroll
    for (int r = 0; r < 3; r++) {
        int j = r * 256 + tid;
        float v = 0.f;
        if (j < PUNIT) {
            int oc = p * PDIM + j / P_OW;
            int ow = j % P_OW;
            v = praw[((size_t)(b * P_OW + ow)) * CAPS + oc];
        }
        vals[r] = v; ss += v * v;
    }
    red[tid] = ss; __syncthreads();
    for (int s = 128; s > 0; s >>= 1) { if (tid < s) red[tid] += red[tid + s]; __syncthreads(); }
    float msq = red[0];
    float factor = msq / ((1.f + msq) * sqrtf(msq));
#pragma unroll
    for (int r = 0; r < 3; r++) {
        int j = r * 256 + tid;
        if (j < PUNIT) ut[(size_t)j * (B_ * NPRIM) + bp] = vals[r] * factor;
    }
}

// ---------------- u_hat[i][j][b][d] (fp16) ----------------
__global__ __launch_bounds__(256) void uhat_kernel(const float* __restrict__ ut,
                                                   const float* __restrict__ rW,
                                                   _Float16* __restrict__ uh)
{
    __shared__ float ub[B_ * NPRIM];   // 512
    __shared__ float wl[ODIM * NPRIM]; // 128
    int i = blockIdx.x, j = blockIdx.y;
    int tid = threadIdx.x;
    for (int f = tid; f < 512; f += 256) ub[f] = ut[(size_t)i * 512 + f];
    if (tid < 128) wl[tid] = rW[((size_t)(i * NOUT + j)) * 128 + tid];
    __syncthreads();
    size_t base = ((size_t)(i * NOUT + j)) * 1024;
#pragma unroll
    for (int r = 0; r < 4; r++) {
        int e = r * 256 + tid;
        int b = e >> 4, d = e & 15;
        float acc = 0.f;
#pragma unroll
        for (int p = 0; p < 8; p++) acc += wl[d * 8 + p] * ub[b * 8 + p];
        uh[base + e] = (_Float16)acc;
    }
}

// ---------------- routing ----------------
__global__ void zero_bij(float* __restrict__ bij)
{
    int idx = blockIdx.x * 256 + threadIdx.x;
    if (idx < PUNIT * NOUT) bij[idx] = 0.f;
}

__global__ __launch_bounds__(128) void softmax_c(const float* __restrict__ bij,
                                                 float* __restrict__ cmat)
{
    __shared__ float red[128];
    int i = blockIdx.x, tid = threadIdx.x;
    float v = (tid < NOUT) ? bij[i * NOUT + tid] : -1e30f;
    red[tid] = v; __syncthreads();
    for (int s = 64; s > 0; s >>= 1) { if (tid < s) red[tid] = fmaxf(red[tid], red[tid + s]); __syncthreads(); }
    float mx = red[0]; __syncthreads();
    float e = (tid < NOUT) ? __expf(v - mx) : 0.f;
    red[tid] = e; __syncthreads();
    for (int s = 64; s > 0; s >>= 1) { if (tid < s) red[tid] += red[tid + s]; __syncthreads(); }
    float sum = red[0];
    if (tid < NOUT) cmat[i * NOUT + tid] = e / sum;
}

__global__ __launch_bounds__(256) void sj_kernel(const float* __restrict__ cmat,
                                                 const _Float16* __restrict__ uh,
                                                 float* __restrict__ smat)
{
    int j = blockIdx.x, bq = blockIdx.y;
    int e = bq * 256 + threadIdx.x;      // (b,d) index in 1024
    float acc = 0.f;
    for (int i = 0; i < PUNIT; i++) {
        float c = cmat[i * NOUT + j];
        acc += c * (float)uh[((size_t)(i * NOUT + j)) * 1024 + e];
    }
    smat[(size_t)j * 1024 + e] = acc;
}

__global__ __launch_bounds__(128) void vj_kernel(const float* __restrict__ smat,
                                                 float* __restrict__ vmat)
{
    __shared__ float fac[ODIM];
    int b = blockIdx.x, tid = threadIdx.x;
    if (tid < ODIM) {
        float ss = 0.f;
        for (int j = 0; j < NOUT; j++) {
            float s = smat[(size_t)j * 1024 + b * 16 + tid]; ss += s * s;
        }
        fac[tid] = ss / ((1.f + ss) * sqrtf(ss));
    }
    __syncthreads();
    for (int e = tid; e < NOUT * ODIM; e += 128) {
        int j = e >> 4, d = e & 15;
        size_t o = (size_t)j * 1024 + b * 16 + d;
        vmat[o] = smat[o] * fac[d];
    }
}

__global__ __launch_bounds__(256) void agree_kernel(const _Float16* __restrict__ uh,
                                                    const float* __restrict__ vmat,
                                                    float* __restrict__ bij)
{
    __shared__ float red[256];
    int i = blockIdx.x, j = blockIdx.y, tid = threadIdx.x;
    size_t ub = ((size_t)(i * NOUT + j)) * 1024;
    size_t vb = (size_t)j * 1024;
    float a = 0.f;
#pragma unroll
    for (int r = 0; r < 4; r++) { int e = r * 256 + tid; a += (float)uh[ub + e] * vmat[vb + e]; }
    red[tid] = a; __syncthreads();
    for (int s = 128; s > 0; s >>= 1) { if (tid < s) red[tid] += red[tid + s]; __syncthreads(); }
    if (tid == 0) bij[i * NOUT + j] += red[0] * (1.f / 64.f);
}

__global__ __launch_bounds__(128) void final_kernel(const float* __restrict__ vmat,
                                                    float* __restrict__ out)
{
    __shared__ float red[128];
    int b = blockIdx.x, tid = threadIdx.x;
    float m = 0.f;
    if (tid < NOUT) {
        float ss = 0.f;
        for (int d = 0; d < ODIM; d++) {
            float v = vmat[(size_t)tid * 1024 + b * 16 + d]; ss += v * v;
        }
        m = sqrtf(ss);
    }
    red[tid] = m; __syncthreads();
    for (int s = 64; s > 0; s >>= 1) { if (tid < s) red[tid] = fmaxf(red[tid], red[tid + s]); __syncthreads(); }
    float mx = red[0];
    if (tid < NOUT) out[b * NOUT + tid] = m / mx;
    float* o2 = out + B_ * NOUT;
    for (int e = tid; e < NOUT * ODIM; e += 128) {
        int jj = e >> 4, d = e & 15;
        o2[((size_t)b * NOUT + jj) * ODIM + d] = vmat[(size_t)jj * 1024 + b * 16 + d];
    }
}

// ---------------- host driver ----------------
extern "C" void kernel_launch(void* const* d_in, const int* in_sizes, int n_in,
                              void* d_out, int out_size, void* d_ws, size_t ws_size,
                              hipStream_t stream)
{
    (void)in_sizes; (void)n_in; (void)out_size; (void)ws_size;
    const float* x      = (const float*)d_in[0];
    const float* conv_w = (const float*)d_in[1];
    const float* conv_b = (const float*)d_in[2];
    const float* w_ih   = (const float*)d_in[3];
    const float* w_hh   = (const float*)d_in[4];
    const float* b_ih   = (const float*)d_in[5];
    const float* b_hh   = (const float*)d_in[6];
    const float* caps_w = (const float*)d_in[7];
    const float* caps_b = (const float*)d_in[8];
    const float* prim_w = (const float*)d_in[9];
    const float* prim_b = (const float*)d_in[10];
    const float* route_W= (const float*)d_in[11];

    char* ws = (char*)d_ws;
    size_t off = 0;
    auto alloc = [&](size_t bytes) -> void* {
        void* p = ws + off;
        off = (off + bytes + 255) & ~(size_t)255;
        return p;
    };
    _Float16* seq16   = (_Float16*)alloc((size_t)NSEQ * WW * C1 * 2);
    _Float16* Wcat    = (_Float16*)alloc((size_t)NGATE * KCAT * 2);
    float*    bsum    = (float*)   alloc((size_t)NGATE * 4);
    _Float16* xcat    = (_Float16*)alloc((size_t)NSEQ * KCAT * 2);
    float*    cst     = (float*)   alloc((size_t)NSEQ * LH * 4);
    float*    gates   = (float*)   alloc((size_t)NSEQ * NGATE * 4);
    _Float16* hs16    = (_Float16*)alloc((size_t)NSEQ * WW * LH * 2);
    _Float16* capsW   = (_Float16*)alloc((size_t)CAPS * C_K * 2);
    _Float16* capsact = (_Float16*)alloc((size_t)C_M * CAPS * 2);
    _Float16* primW   = (_Float16*)alloc((size_t)CAPS * P_K * 2);
    float*    praw    = (float*)   alloc((size_t)P_M * CAPS * 4);
    float*    ut      = (float*)   alloc((size_t)PUNIT * B_ * NPRIM * 4);
    _Float16* uh      = (_Float16*)alloc((size_t)PUNIT * NOUT * B_ * ODIM * 2);
    float*    bij     = (float*)   alloc((size_t)PUNIT * NOUT * 4);
    float*    cmat    = (float*)   alloc((size_t)PUNIT * NOUT * 4);
    float*    smat    = (float*)   alloc((size_t)NOUT * B_ * ODIM * 4);
    float*    vmat    = (float*)   alloc((size_t)NOUT * B_ * ODIM * 4);

    // stage 1: front-end + weight packing
    conv1_pool<<<NSEQ, 256, 0, stream>>>(x, conv_w, conv_b, seq16);
    pack_lstm<<<(NGATE * KCAT + 255) / 256, 256, 0, stream>>>(w_ih, w_hh, b_ih, b_hh, Wcat, bsum);
    pack_conv<<<(CAPS * C_IC * 81 + 255) / 256, 256, 0, stream>>>(caps_w, capsW, CAPS, C_IC, 9, 9);
    pack_conv<<<(CAPS * P_IC * 81 + 255) / 256, 256, 0, stream>>>(prim_w, primW, CAPS, P_IC, 9, 9);

    // stage 2: LSTM over 49 steps (WMMA GEMM per step)
    lstm_init<<<(NSEQ * LH + 255) / 256, 256, 0, stream>>>(cst, xcat);
    for (int t = 0; t < WW; t++) {
        lstm_xcopy<<<(NSEQ * C1 + 255) / 256, 256, 0, stream>>>(seq16, xcat, t);
        gemm_wmma<false><<<dim3((NSEQ + 127) / 128, NGATE / 128), 256, 0, stream>>>(
            xcat, Wcat, gates, nullptr, nullptr, NSEQ, NGATE, KCAT, 0,
            0, 0, 0, 0, 0, 0, 0, 0, 0);
        lstm_gate<<<(NSEQ * LH + 255) / 256, 256, 0, stream>>>(gates, bsum, cst, xcat, hs16, t);
    }

    // stage 3: capsule convolutions (implicit-im2col WMMA)
    gemm_wmma<true><<<dim3((C_M + 127) / 128, CAPS / 128), 256, 0, stream>>>(
        hs16, capsW, nullptr, capsact, caps_b, C_M, CAPS, C_K, 1,
        C_IH, C_IW, C_IC, 9, 9, 1, 1, C_OH, C_OW);
    gemm_wmma<true><<<dim3((P_M + 127) / 128, CAPS / 128), 256, 0, stream>>>(
        capsact, primW, praw, nullptr, prim_b, P_M, CAPS, P_K, 0,
        C_OH, C_OW, P_IC, 9, 9, 2, 2, 1, P_OW);

    // stage 4: squash + u_hat + dynamic routing
    squash_u<<<B_ * NPRIM, 256, 0, stream>>>(praw, ut);
    uhat_kernel<<<dim3(PUNIT, NOUT), 256, 0, stream>>>(ut, route_W, uh);
    zero_bij<<<(PUNIT * NOUT + 255) / 256, 256, 0, stream>>>(bij);
    for (int it = 0; it < 3; it++) {
        softmax_c<<<PUNIT, 128, 0, stream>>>(bij, cmat);
        sj_kernel<<<dim3(NOUT, 4), 256, 0, stream>>>(cmat, uh, smat);
        vj_kernel<<<B_, 128, 0, stream>>>(smat, vmat);
        if (it < 2) agree_kernel<<<dim3(PUNIT, NOUT), 256, 0, stream>>>(uh, vmat, bij);
    }
    final_kernel<<<B_, 128, 0, stream>>>(vmat, (float*)d_out);
}
